// SparseMoE_83803401880038
// MI455X (gfx1250) — matrix operand order repaired and checked by reading, not compiled, not run on hardware
//
#include <hip/hip_runtime.h>
#include <hip/hip_bf16.h>
#include <math.h>

#define TOKENS  8192
#define CDIM    768
#define EXPERTS 8
#define HDIM    3072
#define MT      32            // tokens per block
#define HB      128           // hidden-dim chunk
#define XSTR    (CDIM + 8)    // padded LDS row stride (bf16 elems)
#define HSTR    (HB + 8)
#define KT1     (CDIM / 32)   // 24  k-tiles for GEMM1
#define NT1     (HDIM / 16)   // 192 n-tiles for GEMM1
#define KT2     (HDIM / 32)   // 96  k-tiles for GEMM2
#define NT2     (CDIM / 16)   // 48  n-tiles for GEMM2

typedef __bf16 bf16_t;
typedef __attribute__((ext_vector_type(16))) __bf16 v16bf;
typedef __attribute__((ext_vector_type(8)))  __bf16 v8bf;
typedef __attribute__((ext_vector_type(8)))  float  v8f;

union AFrag { v16bf v; v8bf h[2]; };

__device__ __forceinline__ bf16_t f2bf(float f) {
  union { float f; unsigned u; } a; a.f = f;
  unsigned r = a.u + 0x7FFFu + ((a.u >> 16) & 1u);   // round-to-nearest-even
  union { unsigned short s; bf16_t b; } o; o.s = (unsigned short)(r >> 16);
  return o.b;
}

// ---------------------------------------------------------------- zero
__global__ __launch_bounds__(256) void k_zero(float* __restrict__ y,
                                              int* __restrict__ counts, int ny) {
  int i = blockIdx.x * 256 + threadIdx.x;
  if (i < ny) y[i] = 0.f;
  if (i < EXPERTS) counts[i] = 0;
}

// ---------------------------------------------------------------- x -> bf16
__global__ __launch_bounds__(256) void k_cvt_x(const float* __restrict__ x,
                                               bf16_t* __restrict__ xb, int n) {
  int i = blockIdx.x * 256 + threadIdx.x;
  if (i < n) xb[i] = f2bf(x[i]);
}

// ---------------------------------------------------------------- W1 -> bf16 swizzled B-fragments
// layout: [frag][half(2)][lane(32)][j(8)] ; frag = (e*NT1 + nt)*KT1 + kt
// lane holds col n = nt*16 + (lane&15), rows k = kt*32 + half*16 + (lane<16?0:8) + j
__global__ __launch_bounds__(256) void k_cvt_w1(const float* __restrict__ W1,
                                                bf16_t* __restrict__ W1b) {
  long t = (long)blockIdx.x * 256 + threadIdx.x;
  if (t >= (long)EXPERTS * CDIM * HDIM) return;
  int j = (int)(t & 7); int lane = (int)((t >> 3) & 31); int half = (int)((t >> 8) & 1);
  long frag = t >> 9;
  int kt = (int)(frag % KT1); long rem = frag / KT1;
  int nt = (int)(rem % NT1); int e = (int)(rem / NT1);
  int k = kt * 32 + ((lane & 16) ? 8 : 0) + half * 16 + j;
  int n = nt * 16 + (lane & 15);
  W1b[t] = f2bf(W1[((size_t)e * CDIM + k) * HDIM + n]);
}

__global__ __launch_bounds__(256) void k_cvt_w2(const float* __restrict__ W2,
                                                bf16_t* __restrict__ W2b) {
  long t = (long)blockIdx.x * 256 + threadIdx.x;
  if (t >= (long)EXPERTS * CDIM * HDIM) return;
  int j = (int)(t & 7); int lane = (int)((t >> 3) & 31); int half = (int)((t >> 8) & 1);
  long frag = t >> 9;
  int kt = (int)(frag % KT2); long rem = frag / KT2;
  int nt = (int)(rem % NT2); int e = (int)(rem / NT2);
  int k = kt * 32 + ((lane & 16) ? 8 : 0) + half * 16 + j;
  int n = nt * 16 + (lane & 15);
  W2b[t] = f2bf(W2[((size_t)e * HDIM + k) * CDIM + n]);
}

// ---------------------------------------------------------------- router: one wave per token
__global__ __launch_bounds__(256) void k_router(
    const float* __restrict__ x, const float* __restrict__ noise,
    const float* __restrict__ Wr, const float* __restrict__ br,
    const float* __restrict__ Wn, const float* __restrict__ bn,
    int* __restrict__ counts, int* __restrict__ listTok, float* __restrict__ listGate) {
  int lane = threadIdx.x & 31;
  int wave = threadIdx.x >> 5;
  int tok = blockIdx.x * 8 + wave;
  if (tok >= TOKENS) return;
  float nz[EXPERTS];
  #pragma unroll
  for (int e = 0; e < EXPERTS; ++e) {
    float s = 0.f, sn = 0.f;
    for (int c = lane; c < CDIM; c += 32) {
      float xv = x[(size_t)tok * CDIM + c];
      s  += xv * Wr[c * EXPERTS + e];
      sn += xv * Wn[c * EXPERTS + e];
    }
    #pragma unroll
    for (int off = 16; off >= 1; off >>= 1) {
      s  += __shfl_xor(s, off, 32);
      sn += __shfl_xor(sn, off, 32);
    }
    float lg = s + br[e];
    float sv = sn + bn[e];
    float sp = (sv > 20.f) ? sv : log1pf(expf(sv));     // softplus
    nz[e] = lg + sp * noise[(size_t)tok * EXPERTS + e];
  }
  if (lane == 0) {
    int i0 = 0; float v0 = nz[0];
    #pragma unroll
    for (int e = 1; e < EXPERTS; ++e) if (nz[e] > v0) { v0 = nz[e]; i0 = e; }
    int i1 = (i0 == 0) ? 1 : 0; float v1 = nz[i1];
    #pragma unroll
    for (int e = 0; e < EXPERTS; ++e) {
      if (e == i0 || e == i1) continue;
      if (nz[e] > v1) { v1 = nz[e]; i1 = e; }
    }
    float ex = expf(v1 - v0);                 // v0 >= v1: stable softmax
    float g0 = 1.f / (1.f + ex);
    float g1 = ex / (1.f + ex);
    int p0 = atomicAdd(&counts[i0], 1);
    listTok[i0 * TOKENS + p0] = tok; listGate[i0 * TOKENS + p0] = g0;
    int p1 = atomicAdd(&counts[i1], 1);
    listTok[i1 * TOKENS + p1] = tok; listGate[i1 * TOKENS + p1] = g1;
  }
}

// ---------------------------------------------------------------- fused expert FFN (WMMA)
__global__ __launch_bounds__(256) void k_moe_ffn(
    const bf16_t* __restrict__ xb, const float* __restrict__ x,
    const bf16_t* __restrict__ W1b, const bf16_t* __restrict__ W2b,
    const float* __restrict__ b1, const float* __restrict__ b2,
    const float* __restrict__ gamma, const float* __restrict__ beta,
    const int* __restrict__ counts, const int* __restrict__ listTok,
    const float* __restrict__ listGate, float* __restrict__ y) {
  __shared__ __align__(16) bf16_t x_s[MT * XSTR];
  __shared__ __align__(16) bf16_t h_s[MT * HSTR];
  __shared__ int   tok_s[MT];
  __shared__ float gate_s[MT];
  __shared__ float rsum[MT], rsum2[MT], mu_s[MT], rstd_s[MT];

  const int e    = blockIdx.x;
  const int tile = blockIdx.y;
  const int cnt  = counts[e];
  if (tile * MT >= cnt) return;

  const int tid   = threadIdx.x;
  const int lane  = tid & 31;
  const int w     = tid >> 5;            // wave 0..7
  const int nl    = lane & 15;
  const int half8 = (lane < 16) ? 0 : 8; // C/D row offset for upper half-wave
  const int kb    = (lane < 16) ? 0 : 8; // A-fragment K base

  if (tid < MT) {
    int idx = tile * MT + tid;
    if (idx < cnt) { tok_s[tid] = listTok[e * TOKENS + idx];
                     gate_s[tid] = listGate[e * TOKENS + idx]; }
    else           { tok_s[tid] = -1; gate_s[tid] = 0.f; }
    rsum[tid] = 0.f; rsum2[tid] = 0.f;
  }
  __syncthreads();

  { // stage 32 x-rows (bf16) into LDS, gathered by token
    int r = tid >> 3, tc = tid & 7;
    int tk = tok_s[r]; if (tk < 0) tk = tok_s[0];
    const v8bf* src = (const v8bf*)(xb + (size_t)tk * CDIM);
    #pragma unroll
    for (int i = 0; i < 12; ++i) {
      int cc = i * 8 + tc;
      *(v8bf*)&x_s[r * XSTR + cc * 8] = src[cc];
    }
  }
  __syncthreads();

  const v8f vzero = {};
  v8f acc[12];                           // [mtile 0..1][6 col-tiles] of 16x16 f32
  #pragma unroll
  for (int i = 0; i < 12; ++i) acc[i] = vzero;

  const v8bf* w1p = (const v8bf*)W1b;
  const v8bf* w2p = (const v8bf*)W2b;

  for (int hb = 0; hb < HDIM / HB; ++hb) {
    // ---- GEMM1: h[32,HB] = gelu(x_tile @ W1 + b1); wave w owns h cols [w*16,w*16+16)
    {
      v8f ha0 = vzero, ha1 = vzero;
      const int ntg = hb * (HB / 16) + w;
      const v8bf* bp = w1p + ((size_t)(e * NT1 + ntg)) * KT1 * 64 + lane;
      AFrag bb[2], a0, a1;
      bb[0].h[0] = bp[0];                 // prime pipeline: B(kt=0)
      bb[0].h[1] = bp[32];
      #pragma unroll 4
      for (int kt = 0; kt < KT1; ++kt) {
        // prefetch B(kt+1); last one reads 1 KiB (one fragment) past the tile (still inside ws) - harmless
        bb[(kt + 1) & 1].h[0] = bp[(kt + 1) * 64];
        bb[(kt + 1) & 1].h[1] = bp[(kt + 1) * 64 + 32];
        const int col = kt * 32 + kb;
        a0.h[0] = *(const v8bf*)&x_s[nl * XSTR + col];
        a0.h[1] = *(const v8bf*)&x_s[nl * XSTR + col + 16];
        a1.h[0] = *(const v8bf*)&x_s[(16 + nl) * XSTR + col];
        a1.h[1] = *(const v8bf*)&x_s[(16 + nl) * XSTR + col + 16];
        // keep the prefetch above the WMMAs: nothing may cross this point
        __builtin_amdgcn_sched_barrier(0);
        ha0 = __builtin_amdgcn_wmma_f32_16x16x32_bf16(false, a0.v, false, bb[kt & 1].v, (short)0, ha0, false, false);
        ha1 = __builtin_amdgcn_wmma_f32_16x16x32_bf16(false, a1.v, false, bb[kt & 1].v, (short)0, ha1, false, false);
      }
      const int nloc = w * 16 + nl;
      const float b1v = b1[(size_t)e * HDIM + hb * HB + nloc];
      #pragma unroll
      for (int r = 0; r < 8; ++r) {
        float v0 = ha0[r] + b1v;
        float v1 = ha1[r] + b1v;
        float g0 = 0.5f * v0 * (1.f + erff(v0 * 0.70710678118654752f)); // exact gelu
        float g1 = 0.5f * v1 * (1.f + erff(v1 * 0.70710678118654752f));
        h_s[(r + half8) * HSTR + nloc]      = f2bf(g0);
        h_s[(16 + r + half8) * HSTR + nloc] = f2bf(g1);
      }
    }
    __syncthreads();
    // ---- GEMM2: acc += h_tile @ W2 ; wave w owns out col-tiles [w*6, w*6+6)
    #pragma unroll
    for (int kk = 0; kk < HB / 32; ++kk) {
      AFrag a0, a1;
      const int col = kk * 32 + kb;
      a0.h[0] = *(const v8bf*)&h_s[nl * HSTR + col];
      a0.h[1] = *(const v8bf*)&h_s[nl * HSTR + col + 16];
      a1.h[0] = *(const v8bf*)&h_s[(16 + nl) * HSTR + col];
      a1.h[1] = *(const v8bf*)&h_s[(16 + nl) * HSTR + col + 16];
      const int ktg = hb * (HB / 32) + kk;
      const v8bf* b2p = w2p + ((size_t)(e * NT2 + w * 6)) * KT2 * 64 + (size_t)ktg * 64 + lane;
      AFrag bb[2];
      bb[0].h[0] = b2p[0];               // prime B(t=0)
      bb[0].h[1] = b2p[32];
      #pragma unroll
      for (int t = 0; t < 6; ++t) {
        if (t < 5) {                     // static in the unrolled loop
          bb[(t + 1) & 1].h[0] = b2p[(size_t)(t + 1) * KT2 * 64];
          bb[(t + 1) & 1].h[1] = b2p[(size_t)(t + 1) * KT2 * 64 + 32];
        }
        // keep the prefetch above the WMMAs consuming the previous fragment
        __builtin_amdgcn_sched_barrier(0);
        acc[t]     = __builtin_amdgcn_wmma_f32_16x16x32_bf16(false, a0.v, false, bb[t & 1].v, (short)0, acc[t], false, false);
        acc[6 + t] = __builtin_amdgcn_wmma_f32_16x16x32_bf16(false, a1.v, false, bb[t & 1].v, (short)0, acc[6 + t], false, false);
      }
    }
    __syncthreads();
  }

  // ---- epilogue: +b2, fp32 residual, LayerNorm, gamma/beta, gate, atomic combine
  #pragma unroll
  for (int mt = 0; mt < 2; ++mt) {
    const int mbase = mt * 16 + half8;
    int tks[8];
    #pragma unroll
    for (int r = 0; r < 8; ++r) { int tm = tok_s[mbase + r]; tks[r] = (tm < 0) ? tok_s[0] : tm; }
    float s1[8], s2[8];
    #pragma unroll
    for (int r = 0; r < 8; ++r) { s1[r] = 0.f; s2[r] = 0.f; }
    #pragma unroll
    for (int t = 0; t < 6; ++t) {
      const int n = (w * 6 + t) * 16 + nl;
      const float b2v = b2[(size_t)e * CDIM + n];
      #pragma unroll
      for (int r = 0; r < 8; ++r) {
        float res = acc[mt * 6 + t][r] + b2v + x[(size_t)tks[r] * CDIM + n];
        acc[mt * 6 + t][r] = res;
        s1[r] += res; s2[r] += res * res;
      }
    }
    #pragma unroll
    for (int off = 8; off >= 1; off >>= 1) {   // width-16 keeps half-waves separate
      #pragma unroll
      for (int r = 0; r < 8; ++r) {
        s1[r] += __shfl_xor(s1[r], off, 16);
        s2[r] += __shfl_xor(s2[r], off, 16);
      }
    }
    if (nl == 0) {
      #pragma unroll
      for (int r = 0; r < 8; ++r) {
        atomicAdd(&rsum[mbase + r], s1[r]);
        atomicAdd(&rsum2[mbase + r], s2[r]);
      }
    }
  }
  __syncthreads();
  if (tid < MT) {
    float mu  = rsum[tid]  * (1.f / CDIM);
    float var = rsum2[tid] * (1.f / CDIM) - mu * mu;
    mu_s[tid]   = mu;
    rstd_s[tid] = rsqrtf(var + 1e-5f);
  }
  __syncthreads();
  #pragma unroll
  for (int mt = 0; mt < 2; ++mt) {
    const int mbase = mt * 16 + half8;
    #pragma unroll
    for (int t = 0; t < 6; ++t) {
      const int n = (w * 6 + t) * 16 + nl;
      const float gmv = gamma[(size_t)e * CDIM + n];
      const float btv = beta[(size_t)e * CDIM + n];
      #pragma unroll
      for (int r = 0; r < 8; ++r) {
        const int m = mbase + r;
        const int tm = tok_s[m];
        if (tm >= 0) {
          float o = (acc[mt * 6 + t][r] - mu_s[m]) * rstd_s[m] * gmv + btv;
          atomicAdd(&y[(size_t)tm * CDIM + n], o * gate_s[m]);
        }
      }
    }
  }
}

// ---------------------------------------------------------------- host
extern "C" void kernel_launch(void* const* d_in, const int* in_sizes, int n_in,
                              void* d_out, int out_size, void* d_ws, size_t ws_size,
                              hipStream_t stream) {
  (void)in_sizes; (void)n_in; (void)out_size; (void)ws_size;
  const float* x     = (const float*)d_in[0];
  const float* noise = (const float*)d_in[1];
  const float* Wr    = (const float*)d_in[2];
  const float* br    = (const float*)d_in[3];
  const float* Wn    = (const float*)d_in[4];
  const float* bn    = (const float*)d_in[5];
  const float* W1    = (const float*)d_in[6];
  const float* b1    = (const float*)d_in[7];
  const float* W2    = (const float*)d_in[8];
  const float* b2    = (const float*)d_in[9];
  const float* gamma = (const float*)d_in[10];
  const float* beta  = (const float*)d_in[11];
  float* y = (float*)d_out;

  char* ws = (char*)d_ws;
  size_t off = 0;
  auto take = [&](size_t bytes) -> char* {
    char* p = ws + off;
    off += (bytes + 255) & ~(size_t)255;
    return p;
  };
  bf16_t* xb       = (bf16_t*)take((size_t)TOKENS * CDIM * 2);
  bf16_t* W1b      = (bf16_t*)take((size_t)EXPERTS * CDIM * HDIM * 2);
  bf16_t* W2b      = (bf16_t*)take((size_t)EXPERTS * CDIM * HDIM * 2);
  int*    counts   = (int*)take(EXPERTS * 4);
  int*    listTok  = (int*)take((size_t)EXPERTS * TOKENS * 4);
  float*  listGate = (float*)take((size_t)EXPERTS * TOKENS * 4);

  const int ny = TOKENS * CDIM;
  k_zero <<<(ny + 255) / 256, 256, 0, stream>>>(y, counts, ny);
  k_cvt_x<<<(ny + 255) / 256, 256, 0, stream>>>(x, xb, ny);
  const long nw = (long)EXPERTS * CDIM * HDIM;
  k_cvt_w1<<<(int)((nw + 255) / 256), 256, 0, stream>>>(W1, W1b);
  k_cvt_w2<<<(int)((nw + 255) / 256), 256, 0, stream>>>(W2, W2b);
  k_router<<<TOKENS / 8, 256, 0, stream>>>(x, noise, Wr, br, Wn, bn,
                                           counts, listTok, listGate);
  dim3 grid(EXPERTS, TOKENS / MT);
  k_moe_ffn<<<grid, 256, 0, stream>>>(xb, x, W1b, W2b, b1, b2, gamma, beta,
                                      counts, listTok, listGate, y);
}
